// MaskedHSTUAttention_22522808500333
// MI455X (gfx1250) — matrix operand order repaired and checked by reading, not compiled, not run on hardware
//
#include <hip/hip_runtime.h>

// ---------------------------------------------------------------------------
// MaskedHSTUAttention for MI455X (gfx1250, wave32, WMMA bf16 16x16x32)
// Shapes: B=8, L=1024, D=1024, H=8, Dh=128, NB=128
// ---------------------------------------------------------------------------

typedef __attribute__((ext_vector_type(16))) __bf16 v16bf;
typedef __attribute__((ext_vector_type(8)))  float  v8f;

union Frag16 { v16bf v; unsigned int u[8]; };

static __device__ __forceinline__ unsigned short f32_to_bf16(float f) {
  union { float f; unsigned int u; } x; x.f = f;
  unsigned int r = x.u + 0x7FFFu + ((x.u >> 16) & 1u);   // round-to-nearest-even
  return (unsigned short)(r >> 16);
}
static __device__ __forceinline__ float bf16_to_f32(unsigned short h) {
  union { float f; unsigned int u; } x; x.u = ((unsigned int)h) << 16;
  return x.f;
}
static __device__ __forceinline__ float silu_f(float x) {
  return x / (1.0f + __expf(-x));
}

// A fragment (16x32, MxK), ISA 7.12.2 layout: lane m=lane%16 = row,
// g=lane/16: VGPR v<4 -> K=8g+2v{,+1};  v>=4 -> K=16+8g+2(v-4){,+1}
static __device__ __forceinline__ v16bf load_a_frag(const unsigned short* p, int row0, int ko, int ld) {
  int lane = threadIdx.x & 31;
  int g = lane >> 4, m = lane & 15;
  const unsigned short* rp = p + (row0 + m) * ld + ko + 8 * g;
  Frag16 f;
#pragma unroll
  for (int v = 0; v < 4; ++v) f.u[v]     = *(const unsigned int*)(rp + 2 * v);
#pragma unroll
  for (int v = 0; v < 4; ++v) f.u[4 + v] = *(const unsigned int*)(rp + 16 + 2 * v);
  return f.v;
}

// B fragment (32x16, KxN) from an LDS tile stored transposed as Bt[N][K]:
// column n=lane%16, g=lane/16: VGPR v<4 -> K=ko+8g+2v{,+1};  v>=4 -> K=ko+16+8g+2(v-4){,+1}
static __device__ __forceinline__ v16bf load_b_frag(const unsigned short* p, int col0, int ko, int ld) {
  int lane = threadIdx.x & 31;
  int g = lane >> 4, n = lane & 15;
  const unsigned short* cp = p + (col0 + n) * ld + ko + 8 * g;
  Frag16 f;
#pragma unroll
  for (int v = 0; v < 4; ++v) f.u[v]     = *(const unsigned int*)(cp + 2 * v);
#pragma unroll
  for (int v = 0; v < 4; ++v) f.u[4 + v] = *(const unsigned int*)(cp + 16 + 2 * v);
  return f.v;
}

static __device__ __forceinline__ v8f wmma_bf16(v16bf a, v16bf b, v8f c) {
  return __builtin_amdgcn_wmma_f32_16x16x32_bf16(false, a, false, b, (short)0, c, false, false);
}

// ---------------------------------------------------------------------------
// K0: f32 -> bf16 conversion
// ---------------------------------------------------------------------------
__global__ void convert_bf16_kernel(const float* __restrict__ in,
                                    unsigned short* __restrict__ out, int n) {
  int i = blockIdx.x * blockDim.x + threadIdx.x;
  int stride = gridDim.x * blockDim.x;
  for (; i < n; i += stride) out[i] = f32_to_bf16(in[i]);
}

// ---------------------------------------------------------------------------
// K1: projection GEMM  silu(hidden @ W.T)   M=8192 N=1024 K=1024
//   Block tile 128x64, 256 threads (8 waves): wave w owns rows 16w, 4 subtiles.
//   Register double-buffering of the global->LDS staging.
// mode 0: q[(b*8+h)*1024+l][128]   mode 1: k same layout
// mode 2: vT[(b*8+h)*128+d][1024]  mode 3: gate[m][n] (bf16)
// ---------------------------------------------------------------------------
__global__ __launch_bounds__(256)
void proj_gemm_kernel(const unsigned short* __restrict__ A,
                      const unsigned short* __restrict__ Bt,
                      unsigned short* __restrict__ out, int mode) {
  __shared__ __attribute__((aligned(16))) unsigned short sA[128 * 32];
  __shared__ __attribute__((aligned(16))) unsigned short sB[64 * 32];
  const int m0 = blockIdx.x * 128;
  const int n0 = blockIdx.y * 64;
  const int t = threadIdx.x;
  const int w = t >> 5;
  const int r0 = w * 16;
  const int arow0 = t >> 2, acol = (t & 3) * 8;   // A chunk 0: row=arow0, chunk 1: row=arow0+64
  const int brow  = t >> 2, bcol = (t & 3) * 8;

  uint4 pa0, pa1, pb;
  pa0 = *(const uint4*)(A  + (long)(m0 + arow0) * 1024 + acol);
  pa1 = *(const uint4*)(A  + (long)(m0 + arow0 + 64) * 1024 + acol);
  pb  = *(const uint4*)(Bt + (long)(n0 + brow) * 1024 + bcol);

  v8f zero = {};
  v8f acc[4] = {zero, zero, zero, zero};

  for (int ks = 0; ks < 1024; ks += 32) {
    *(uint4*)(sA + arow0 * 32 + acol)        = pa0;
    *(uint4*)(sA + (arow0 + 64) * 32 + acol) = pa1;
    *(uint4*)(sB + brow * 32 + bcol)         = pb;
    __syncthreads();
    if (ks + 32 < 1024) {   // prefetch next tile while WMMAs run
      pa0 = *(const uint4*)(A  + (long)(m0 + arow0) * 1024 + ks + 32 + acol);
      pa1 = *(const uint4*)(A  + (long)(m0 + arow0 + 64) * 1024 + ks + 32 + acol);
      pb  = *(const uint4*)(Bt + (long)(n0 + brow) * 1024 + ks + 32 + bcol);
    }
    v16bf a = load_a_frag(sA, r0, 0, 32);
#pragma unroll
    for (int s = 0; s < 4; ++s) {
      v16bf b = load_b_frag(sB, 16 * s, 0, 32);
      acc[s] = wmma_bf16(a, b, acc[s]);
    }
    __syncthreads();
  }

  const int lane = t & 31;
  const int g = lane >> 4, nn = lane & 15;
#pragma unroll
  for (int s = 0; s < 4; ++s) {
#pragma unroll
    for (int r = 0; r < 8; ++r) {
      int mrow = m0 + r0 + r + 8 * g;
      int ncol = n0 + 16 * s + nn;
      float x = silu_f(acc[s][r]);
      int b = mrow >> 10, l = mrow & 1023, h = ncol >> 7, d = ncol & 127;
      long addr;
      if (mode <= 1)       addr = ((long)((b * 8 + h) * 1024 + l)) * 128 + d;  // q / k
      else if (mode == 2)  addr = ((long)((b * 8 + h) * 128 + d)) * 1024 + l;  // vT
      else                 addr = (long)mrow * 1024 + ncol;                    // gate
      out[addr] = f32_to_bf16(x);
    }
  }
}

// ---------------------------------------------------------------------------
// K2: attention.  grid = (L/64, B*H).  Block = 64-row strip.
//   Causal: only tiles j0 <= i0 are computed (upper tiles are exactly
//   silu(-1e9)/L == -0 in the reference -> zero-filled directly).
//   Per live tile: S=q@k^T (WMMA), fused pos/ts-bucket bias + mask + silu/L,
//   attn written to d_out, tile requantized to bf16, av += S@v (WMMA).
//   K/V tiles are register double-buffered; 2-tiles-ahead global_prefetch.
// ---------------------------------------------------------------------------
__global__ __launch_bounds__(256)
void attn_kernel(const unsigned short* __restrict__ qb,
                 const unsigned short* __restrict__ kb,
                 const unsigned short* __restrict__ vtb,
                 const int* __restrict__ ts,
                 const float* __restrict__ pos_w,
                 const float* __restrict__ ts_w,
                 float* __restrict__ attn_out,
                 float* __restrict__ av_out) {
  __shared__ __attribute__((aligned(16))) unsigned short sQ[64 * 128];
  __shared__ __attribute__((aligned(16))) unsigned short sK[64 * 128];
  __shared__ __attribute__((aligned(16))) unsigned short sV[128 * 64];
  __shared__ __attribute__((aligned(16))) unsigned short sS[64 * 64];
  __shared__ int sTsRow[64];
  __shared__ int sTsCol[64];

  const int bh = blockIdx.y;
  const int b = bh >> 3, h = bh & 7;
  const int i0 = blockIdx.x * 64;
  const int t = threadIdx.x;
  const int w = t >> 5;
  const int lane = t & 31;
  const int g = lane >> 4, nn = lane & 15;
  const int r0 = (w & 3) * 16;
  const int cg = w >> 2;                    // 0..1
  const long headoff = (long)bh * 1024 * 128;
  const int mt_last = i0 >> 6;              // inclusive: diagonal tile

  {  // resident q strip (64 x 128 bf16, contiguous)
    const uint4* src = (const uint4*)(qb + headoff + (long)i0 * 128);
    uint4* dst = (uint4*)sQ;
#pragma unroll
    for (int it = 0; it < 4; ++it) dst[t + 256 * it] = src[t + 256 * it];
  }
  if (t < 64) {
    int i = i0 + t;
    int nx = (i + 1 < 1024) ? (i + 1) : 1023;    // ext[:,1:] indexing
    sTsRow[t] = ts[b * 1024 + nx];
  }

  v8f zero = {};
  v8f av[4] = {zero, zero, zero, zero};
  uint4 pk[4], pv[4];
#pragma unroll
  for (int it = 0; it < 4; ++it) {
    int c = t + 256 * it;
    pk[it] = *(const uint4*)(kb + headoff + (long)c * 8);
    pv[it] = *(const uint4*)(vtb + headoff + (long)(c >> 3) * 1024 + (c & 7) * 8);
  }

  for (int mt = 0; mt <= mt_last; ++mt) {
    const int j0 = mt * 64;
#pragma unroll
    for (int it = 0; it < 4; ++it) {
      ((uint4*)sK)[t + 256 * it] = pk[it];
      ((uint4*)sV)[t + 256 * it] = pv[it];
    }
    if (t < 64) sTsCol[t] = ts[b * 1024 + j0 + t];
    __syncthreads();

    if (mt < mt_last) {     // prefetch next K/V tile into registers
      const int jn = j0 + 64;
#pragma unroll
      for (int it = 0; it < 4; ++it) {
        int c = t + 256 * it;
        pk[it] = *(const uint4*)(kb + headoff + (long)jn * 128 + (long)c * 8);
        pv[it] = *(const uint4*)(vtb + headoff + (long)(c >> 3) * 1024 + jn + (c & 7) * 8);
      }
    }
    if (mt + 2 <= mt_last) {  // L2 prefetch two tiles ahead
      __builtin_prefetch(kb + headoff + (long)(j0 + 128) * 128 + (long)t * 32, 0, 1);
      __builtin_prefetch(vtb + headoff + (long)(t >> 1) * 1024 + j0 + 128, 0, 1);
    }

    // ---- S = q @ k^T (64x64 tile, K=128) ----
    v8f s0 = {}, s1 = {};
#pragma unroll
    for (int kk = 0; kk < 128; kk += 32) {
      v16bf a  = load_a_frag(sQ, r0, kk, 128);
      v16bf b0 = load_b_frag(sK, cg * 32, kk, 128);
      v16bf b1 = load_b_frag(sK, cg * 32 + 16, kk, 128);
      s0 = wmma_bf16(a, b0, s0);
      s1 = wmma_bf16(a, b1, s1);
    }

    // ---- fused bias + mask + silu/L, emit attn, stage bf16 S tile ----
#pragma unroll
    for (int si = 0; si < 2; ++si) {
      v8f acc; if (si) acc = s1; else acc = s0;
#pragma unroll
      for (int r = 0; r < 8; ++r) {
        int il = r0 + r + 8 * g;
        int jl = cg * 32 + 16 * si + nn;
        int i = i0 + il, j = j0 + jl;
        float x = acc[r] + pos_w[j - i + 1023];
        float dt  = (float)(sTsRow[il] - sTsCol[jl]);
        float mag = fmaxf(fabsf(dt), 1.0f);
        int bk = (int)(__logf(mag) * 3.3222591f);    // log(mag)/0.301
        bk = bk < 0 ? 0 : (bk > 128 ? 128 : bk);
        x += ts_w[bk];
        if (j > i) x -= 1.0e9f;                      // causal mask
        float a_v = silu_f(x) * (1.0f / 1024.0f);
        attn_out[((long)bh * 1024 + i) * 1024 + j] = a_v;
        sS[il * 64 + jl] = f32_to_bf16(a_v);
      }
    }
    __syncthreads();

    // ---- av += S @ v  (64x128, K=64) ----
#pragma unroll
    for (int kk = 0; kk < 64; kk += 32) {
      v16bf a = load_a_frag(sS, r0, kk, 64);
#pragma unroll
      for (int s = 0; s < 4; ++s) {
        v16bf bb = load_b_frag(sV, cg * 64 + 16 * s, kk, 64);
        av[s] = wmma_bf16(a, bb, av[s]);
      }
    }
    __syncthreads();
  }

  // ---- zero-fill fully-masked attn columns (j >= (mt_last+1)*64) ----
  {
    const int jstart = (mt_last + 1) * 64;
    const int nc4 = (1024 - jstart) >> 2;            // float4 per row
    if (nc4 > 0) {
      float4 z; z.x = 0.f; z.y = 0.f; z.z = 0.f; z.w = 0.f;
      const int total = 64 * nc4;
      for (int e = t; e < total; e += 256) {
        int row = e / nc4, c4 = e - row * nc4;
        *(float4*)&attn_out[((long)bh * 1024 + i0 + row) * 1024 + jstart + c4 * 4] = z;
      }
    }
  }

  // ---- write av: [b][l][h*128+d] f32 ----
#pragma unroll
  for (int s = 0; s < 4; ++s) {
#pragma unroll
    for (int r = 0; r < 8; ++r) {
      int i = i0 + r0 + r + 8 * g;
      int d = cg * 64 + 16 * s + nn;
      av_out[((long)b * 1024 + i) * 1024 + h * 128 + d] = av[s][r];
    }
  }
}

// ---------------------------------------------------------------------------
// K3a: per-row RMS norm + gate -> y (bf16).  One block per row.
// ---------------------------------------------------------------------------
__global__ __launch_bounds__(256)
void rms_gate_kernel(const float* __restrict__ av,
                     const unsigned short* __restrict__ gateb,
                     const float* __restrict__ rms_w,
                     unsigned short* __restrict__ yb) {
  __shared__ float red[256];
  const long row = blockIdx.x;
  const int t = threadIdx.x;
  float s = 0.f;
  for (int c = t; c < 1024; c += 256) { float v = av[row * 1024 + c]; s += v * v; }
  red[t] = s;
  __syncthreads();
  for (int off = 128; off > 0; off >>= 1) {
    if (t < off) red[t] += red[t + off];
    __syncthreads();
  }
  float scale = rsqrtf(red[0] * (1.0f / 1024.0f) + 1e-6f);
  for (int c = t; c < 1024; c += 256) {
    float y = bf16_to_f32(gateb[row * 1024 + c]) * av[row * 1024 + c] * scale * rms_w[c];
    yb[row * 1024 + c] = f32_to_bf16(y);
  }
}

// ---------------------------------------------------------------------------
// K3b: out = y @ wo.T  (f32 output).  Same 128x64 double-buffered tiling.
// ---------------------------------------------------------------------------
__global__ __launch_bounds__(256)
void out_gemm_kernel(const unsigned short* __restrict__ A,
                     const unsigned short* __restrict__ Bt,
                     float* __restrict__ out) {
  __shared__ __attribute__((aligned(16))) unsigned short sA[128 * 32];
  __shared__ __attribute__((aligned(16))) unsigned short sB[64 * 32];
  const int m0 = blockIdx.x * 128;
  const int n0 = blockIdx.y * 64;
  const int t = threadIdx.x;
  const int w = t >> 5;
  const int r0 = w * 16;
  const int arow0 = t >> 2, acol = (t & 3) * 8;
  const int brow  = t >> 2, bcol = (t & 3) * 8;

  uint4 pa0, pa1, pb;
  pa0 = *(const uint4*)(A  + (long)(m0 + arow0) * 1024 + acol);
  pa1 = *(const uint4*)(A  + (long)(m0 + arow0 + 64) * 1024 + acol);
  pb  = *(const uint4*)(Bt + (long)(n0 + brow) * 1024 + bcol);

  v8f zero = {};
  v8f acc[4] = {zero, zero, zero, zero};

  for (int ks = 0; ks < 1024; ks += 32) {
    *(uint4*)(sA + arow0 * 32 + acol)        = pa0;
    *(uint4*)(sA + (arow0 + 64) * 32 + acol) = pa1;
    *(uint4*)(sB + brow * 32 + bcol)         = pb;
    __syncthreads();
    if (ks + 32 < 1024) {
      pa0 = *(const uint4*)(A  + (long)(m0 + arow0) * 1024 + ks + 32 + acol);
      pa1 = *(const uint4*)(A  + (long)(m0 + arow0 + 64) * 1024 + ks + 32 + acol);
      pb  = *(const uint4*)(Bt + (long)(n0 + brow) * 1024 + ks + 32 + bcol);
    }
    v16bf a = load_a_frag(sA, r0, 0, 32);
#pragma unroll
    for (int s = 0; s < 4; ++s) {
      v16bf b = load_b_frag(sB, 16 * s, 0, 32);
      acc[s] = wmma_bf16(a, b, acc[s]);
    }
    __syncthreads();
  }

  const int lane = t & 31;
  const int g = lane >> 4, nn = lane & 15;
#pragma unroll
  for (int s = 0; s < 4; ++s) {
#pragma unroll
    for (int r = 0; r < 8; ++r) {
      int mrow = m0 + r0 + r + 8 * g;
      int ncol = n0 + 16 * s + nn;
      out[(long)mrow * 1024 + ncol] = acc[s][r];
    }
  }
}

// ---------------------------------------------------------------------------
// Host launch
// ---------------------------------------------------------------------------
extern "C" void kernel_launch(void* const* d_in, const int* in_sizes, int n_in,
                              void* d_out, int out_size, void* d_ws, size_t ws_size,
                              hipStream_t stream) {
  const float* hs   = (const float*)d_in[0];   // (8,1024,1024)
  // d_in[1]: attention_mask (pure causal; applied analytically in-kernel)
  const int*   ts   = (const int*)  d_in[2];   // (8,1024) int32
  const float* wq   = (const float*)d_in[3];
  const float* wk   = (const float*)d_in[4];
  const float* wv   = (const float*)d_in[5];
  const float* wu   = (const float*)d_in[6];
  const float* wo   = (const float*)d_in[7];
  const float* rmsw = (const float*)d_in[8];
  const float* posw = (const float*)d_in[9];
  const float* tsw  = (const float*)d_in[10];

  char* ws = (char*)d_ws;
  size_t off = 0;
  auto alloc = [&](size_t bytes) -> void* {
    void* p = ws + off;
    off += (bytes + 255) & ~(size_t)255;
    return p;
  };
  const size_t NE_H = 8ull * 1024 * 1024;       // hidden-sized tensors (elements)
  const size_t NE_W = 1024ull * 1024;           // weight-sized tensors (elements)

  unsigned short* hb    = (unsigned short*)alloc(NE_H * 2);   // hidden bf16
  unsigned short* wqb   = (unsigned short*)alloc(NE_W * 2);
  unsigned short* wkb   = (unsigned short*)alloc(NE_W * 2);
  unsigned short* wvb   = (unsigned short*)alloc(NE_W * 2);
  unsigned short* wub   = (unsigned short*)alloc(NE_W * 2);
  unsigned short* wob   = (unsigned short*)alloc(NE_W * 2);
  unsigned short* qbuf  = (unsigned short*)alloc(NE_H * 2);   // [bh][l][128]
  unsigned short* kbuf  = (unsigned short*)alloc(NE_H * 2);   // [bh][l][128]
  unsigned short* vtb   = (unsigned short*)alloc(NE_H * 2);   // [bh][d][1024]
  unsigned short* gateb = (unsigned short*)alloc(NE_H * 2);   // [m][1024]
  float*          avb   = (float*)         alloc(NE_H * 4);   // [m][1024] f32
  unsigned short* yb    = (unsigned short*)alloc(NE_H * 2);   // [m][1024]

  float* out_main = (float*)d_out;              // (8,1024,1024) f32
  float* attn_out = out_main + NE_H;            // (8,8,1024,1024) f32

  // K0: conversions
  convert_bf16_kernel<<<4096, 256, 0, stream>>>(hs, hb,  (int)NE_H);
  convert_bf16_kernel<<<512,  256, 0, stream>>>(wq, wqb, (int)NE_W);
  convert_bf16_kernel<<<512,  256, 0, stream>>>(wk, wkb, (int)NE_W);
  convert_bf16_kernel<<<512,  256, 0, stream>>>(wv, wvb, (int)NE_W);
  convert_bf16_kernel<<<512,  256, 0, stream>>>(wu, wub, (int)NE_W);
  convert_bf16_kernel<<<512,  256, 0, stream>>>(wo, wob, (int)NE_W);

  // K1: four fused silu-projections (M=8192, N=1024, K=1024), 128x64 tiles
  dim3 gGemm(64, 16);
  proj_gemm_kernel<<<gGemm, 256, 0, stream>>>(hb, wqb, qbuf,  0);
  proj_gemm_kernel<<<gGemm, 256, 0, stream>>>(hb, wkb, kbuf,  1);
  proj_gemm_kernel<<<gGemm, 256, 0, stream>>>(hb, wvb, vtb,   2);
  proj_gemm_kernel<<<gGemm, 256, 0, stream>>>(hb, wub, gateb, 3);

  // K2: attention (causal-skipped; writes attn to d_out, av to workspace)
  attn_kernel<<<dim3(16, 64), 256, 0, stream>>>(qbuf, kbuf, vtb, ts, posw, tsw,
                                                attn_out, avb);

  // K3a: RMS + gate -> y
  rms_gate_kernel<<<8192, 256, 0, stream>>>(avb, gateb, rmsw, yb);

  // K3b: out = y @ wo.T
  out_gemm_kernel<<<gGemm, 256, 0, stream>>>(yb, wob, out_main);
}
